// TimeAwareMultiHeadSelfAttention_33818572489251
// MI455X (gfx1250) — matrix operand order repaired and checked by reading, not compiled, not run on hardware
//
#include <hip/hip_runtime.h>
#include <hip/hip_bf16.h>
#include <math.h>

// ---------------------------------------------------------------------------
// TimeAwareMultiHeadSelfAttention for MI455X (gfx1250, wave32, WMMA).
// Roofline: dominated by one-pass streaming of time_mat_k/time_mat_v
// (2 x 268 MB f32, ~23 us at 23.3 TB/s). Streaming kernels are unrolled x4
// for memory-level parallelism; padded tail rows are skipped via a per-block
// "first padded index" (mask is arange >= valid, so padding is contiguous).
// All GEMM-shaped math uses v_wmma_f32_16x16x32_f16 (f16 in, f32 accum).
// ---------------------------------------------------------------------------

#define BB 8
#define LL 256
#define DD 128
#define HH 2
#define HD 64
#define SCALE 0.125f   // 1/sqrt(64)

typedef __attribute__((ext_vector_type(16))) _Float16 v16h;
typedef __attribute__((ext_vector_type(8)))  _Float16 v8h;
typedef __attribute__((ext_vector_type(8)))  float    v8f;

__device__ __forceinline__ v16h combine16(v8h lo, v8h hi) {
  v16h r;
#pragma unroll
  for (int i = 0; i < 8; ++i) { r[i] = lo[i]; r[i + 8] = hi[i]; }
  return r;
}

// First padded index in batch row b (padding is a contiguous tail).
__device__ __forceinline__ int first_padded(const unsigned char* mask, int b,
                                            int tid, int* nvsh) {
  if (tid == 0) *nvsh = LL;
  __syncthreads();
  if (mask[b * LL + tid]) atomicMin(nvsh, tid);
  __syncthreads();
  return *nvsh;
}

// ---------------------------------------------------------------------------
// Generic 16-row-tile GEMM: out[M x 128] = X[M x 128] @ W[128 x 128] + bias
// (+ optional f32 "extra" matrix). f16 WMMA, f32 accumulate.
// Block: 256 threads = 8 waves; wave w owns N-tile w (cols 16w..16w+15).
// grid.x = M/16.
// ---------------------------------------------------------------------------
__global__ __launch_bounds__(256) void gemm_proj(
    const float* __restrict__ X, const float* __restrict__ W,
    const float* __restrict__ bias, const float* __restrict__ extra,
    float* __restrict__ outF, _Float16* __restrict__ outH) {
  __shared__ _Float16 xs[16 * 136];     // X tile, row stride 136 (16B-aligned, pad)
  __shared__ _Float16 wts[128 * 136];   // W^T: wts[n][k] = W[k][n]
  const int tid = threadIdx.x;
  const int row0 = blockIdx.x * 16;

  // Stage W^T (coalesced global reads; padded LDS rows avoid bank conflicts).
  for (int idx = tid; idx < 128 * 128; idx += 256) {
    int k = idx >> 7, n = idx & 127;
    wts[n * 136 + k] = (_Float16)W[k * 128 + n];
  }
  // Stage X rows (f32 -> f16).
  for (int idx = tid; idx < 16 * 128; idx += 256) {
    int r = idx >> 7, c = idx & 127;
    xs[r * 136 + c] = (_Float16)X[(row0 + r) * 128 + c];
  }
  __syncthreads();

  const int wave = tid >> 5, lane = tid & 31;
  const int m16 = lane & 15;
  const int abk = (lane & 16) ? 8 : 0;    // A frag K base (ISA 7.12.2)
  const int bbk = (lane & 16) ? 8 : 0;    // B frag K base
  v8f acc = {};
#pragma unroll
  for (int kk = 0; kk < 4; ++kk) {        // K = 128 = 4 x 32
    const _Float16* ap = &xs[m16 * 136 + kk * 32 + abk];
    v16h a = combine16(*(const v8h*)ap, *(const v8h*)(ap + 16));
    const _Float16* bp = &wts[(wave * 16 + m16) * 136 + kk * 32 + bbk];
    v16h bf = combine16(*(const v8h*)bp, *(const v8h*)(bp + 16));
    acc = __builtin_amdgcn_wmma_f32_16x16x32_f16(false, a, false, bf,
                                                 (short)0, acc, false, false);
  }
  const int nglob = wave * 16 + m16;
  const int mbase = (lane >> 4) << 3;     // C/D layout: VGPR r -> M=r / r+8
  const float bv = bias[nglob];
  float vals[8];
#pragma unroll
  for (int r = 0; r < 8; ++r) vals[r] = acc[r] + bv;
  if (extra) {
#pragma unroll
    for (int r = 0; r < 8; ++r)
      vals[r] += extra[(row0 + mbase + r) * 128 + nglob];
  }
  if (outF) {
#pragma unroll
    for (int r = 0; r < 8; ++r)
      outF[(row0 + mbase + r) * 128 + nglob] = vals[r];
  }
  if (outH) {
#pragma unroll
    for (int r = 0; r < 8; ++r)
      outH[(row0 + mbase + r) * 128 + nglob] = (_Float16)vals[r];
  }
}

// ---------------------------------------------------------------------------
// Scores (dense part): wF[b,h,i,j] = sum_d q[b,h,i,d] * (k+pk)[b,h,j,d].
// Block per (b,h,i-tile): grid = 8*2*16 = 256; wave owns 2 j-tiles; K=64.
// ---------------------------------------------------------------------------
__global__ __launch_bounds__(256) void scores_qk(
    const _Float16* __restrict__ qH, const _Float16* __restrict__ kpH,
    float* __restrict__ wF) {
  __shared__ _Float16 qs[16 * 72];
  __shared__ _Float16 ks[256 * 72];
  const int tid = threadIdx.x;
  const int bid = blockIdx.x;
  const int b = bid >> 5, h = (bid >> 4) & 1, i0 = (bid & 15) * 16;

  for (int idx = tid; idx < 16 * 64; idx += 256) {
    int r = idx >> 6, c = idx & 63;
    qs[r * 72 + c] = qH[(b * LL + i0 + r) * DD + h * HD + c];
  }
  for (int idx = tid; idx < 256 * 64; idx += 256) {
    int j = idx >> 6, c = idx & 63;
    ks[j * 72 + c] = kpH[(b * LL + j) * DD + h * HD + c];
  }
  __syncthreads();

  const int wave = tid >> 5, lane = tid & 31;
  const int m16 = lane & 15;
  const int abk = (lane & 16) ? 8 : 0;
  const int bbk = (lane & 16) ? 8 : 0;
#pragma unroll
  for (int t = 0; t < 2; ++t) {
    const int jt = wave * 2 + t;
    v8f acc = {};
#pragma unroll
    for (int kk = 0; kk < 2; ++kk) {      // K = 64
      const _Float16* ap = &qs[m16 * 72 + kk * 32 + abk];
      v16h a = combine16(*(const v8h*)ap, *(const v8h*)(ap + 16));
      const _Float16* bp = &ks[(jt * 16 + m16) * 72 + kk * 32 + bbk];
      v16h bf = combine16(*(const v8h*)bp, *(const v8h*)(bp + 16));
      acc = __builtin_amdgcn_wmma_f32_16x16x32_f16(false, a, false, bf,
                                                   (short)0, acc, false, false);
    }
#pragma unroll
    for (int r = 0; r < 8; ++r) {
      int m = r + ((lane >> 4) << 3);
      wF[((b * HH + h) * LL + i0 + m) * LL + jt * 16 + m16] = acc[r];
    }
  }
}

// ---------------------------------------------------------------------------
// Scores (time part, the bandwidth monster): wF[b,h,i,j] += q[b,h,i,:].tk.
// Block per (b,i); wave owns j-chunks of 4 (4 independent b128 loads in
// flight). Valid-key tail bound `nv` replaces per-j mask byte checks.
// ---------------------------------------------------------------------------
__global__ __launch_bounds__(256) void time_scores(
    const float* __restrict__ qF, const float* __restrict__ tmk,
    const unsigned char* __restrict__ mask, float* __restrict__ wF) {
  __shared__ int nvsh;
  const int b = blockIdx.x / LL, i = blockIdx.x % LL;
  const int tid = threadIdx.x, wave = tid >> 5, lane = tid & 31;
  if (mask[b * LL + i]) return;                       // padded query: zeroed later
  const int nv = first_padded(mask, b, tid, &nvsh);
  const float4 qv = *(const float4*)(qF + (b * LL + i) * DD + lane * 4);
  const long rowbase = ((long)(b * LL + i)) * LL;
  const long h0 = ((long)((b * HH + 0) * LL + i)) * LL;
  const long h1 = ((long)((b * HH + 1) * LL + i)) * LL;
  for (int jb = wave * 4; jb < nv; jb += 32) {
    float4 t[4];
#pragma unroll
    for (int u = 0; u < 4; ++u)           // rows past nv exist in memory: safe
      t[u] = *(const float4*)(tmk + (rowbase + jb + u) * DD + lane * 4);
    float p[4];
#pragma unroll
    for (int u = 0; u < 4; ++u) {
      float s = qv.x * t[u].x + qv.y * t[u].y + qv.z * t[u].z + qv.w * t[u].w;
      s += __shfl_xor(s, 1);
      s += __shfl_xor(s, 2);
      s += __shfl_xor(s, 4);
      s += __shfl_xor(s, 8);              // reduce within 16-lane head halves
      p[u] = s;
    }
    if (lane == 0) {
#pragma unroll
      for (int u = 0; u < 4; ++u)
        if (jb + u < nv) wF[h0 + jb + u] += p[u];
    }
    if (lane == 16) {
#pragma unroll
      for (int u = 0; u < 4; ++u)
        if (jb + u < nv) wF[h1 + jb + u] += p[u];
    }
  }
}

// ---------------------------------------------------------------------------
// Masked softmax over j; writes f32 (for time_out) and f16 (for WMMA).
// Block per (b,h,i) row; grid = 8*2*256 = 4096.
// ---------------------------------------------------------------------------
__global__ __launch_bounds__(256) void softmax_row(
    const unsigned char* __restrict__ mask, float* __restrict__ wF,
    _Float16* __restrict__ wH) {
  __shared__ float red[256];
  const int tid = threadIdx.x;
  const int b = blockIdx.x >> 9, h = (blockIdx.x >> 8) & 1, i = blockIdx.x & 255;
  const long off = ((long)((b * HH + h) * LL + i)) * LL;
  if (mask[b * LL + i]) {                             // padded query -> all-zero row
    wF[off + tid] = 0.f;
    wH[off + tid] = (_Float16)0.f;
    return;
  }
  float s = wF[off + tid] * SCALE;
  if (mask[b * LL + tid]) s = -__builtin_inff();
  red[tid] = s;
  __syncthreads();
  for (int st = 128; st > 0; st >>= 1) {
    if (tid < st) red[tid] = fmaxf(red[tid], red[tid + st]);
    __syncthreads();
  }
  const float mx = red[0];
  __syncthreads();
  const float e = __expf(s - mx);
  red[tid] = e;
  __syncthreads();
  for (int st = 128; st > 0; st >>= 1) {
    if (tid < st) red[tid] += red[tid + st];
    __syncthreads();
  }
  const float r = e / red[0];
  wF[off + tid] = r;
  wH[off + tid] = (_Float16)r;
}

// ---------------------------------------------------------------------------
// attn (dense part): attn[b,i,h*64+d] = sum_j w[b,h,i,j] * (v+pv)[b,h,j,d].
// Block per (b,h,half-of-i); grid = 32; wave owns 4 tiles of 16x16; K=256.
// A frags read straight from global wH (16B-aligned segments).
// ---------------------------------------------------------------------------
__global__ __launch_bounds__(256) void attn_v(
    const _Float16* __restrict__ wH, const _Float16* __restrict__ vpH,
    float* __restrict__ attn) {
  __shared__ _Float16 vpt[64 * 264];                  // vpt[d][j], padded rows
  const int tid = threadIdx.x;
  const int b = blockIdx.x >> 2, h = (blockIdx.x >> 1) & 1, half = blockIdx.x & 1;
  for (int idx = tid; idx < 64 * 256; idx += 256) {
    int d = idx >> 8, j = idx & 255;
    vpt[d * 264 + j] = vpH[(b * LL + j) * DD + h * HD + d];
  }
  __syncthreads();
  const int wave = tid >> 5, lane = tid & 31;
  const int m16 = lane & 15;
  const int abk = (lane & 16) ? 8 : 0;
  const int bbk = (lane & 16) ? 8 : 0;
#pragma unroll
  for (int t = 0; t < 4; ++t) {
    const int tile = wave * 4 + t;
    const int i0 = half * 128 + (tile >> 2) * 16;
    const int nt = tile & 3;
    v8f acc = {};
#pragma unroll
    for (int kk = 0; kk < 8; ++kk) {      // K = 256
      const _Float16* ap =
          &wH[((long)((b * HH + h) * LL + i0 + m16)) * LL + kk * 32 + abk];
      v16h a = combine16(*(const v8h*)ap, *(const v8h*)(ap + 16));
      const _Float16* bp = &vpt[(nt * 16 + m16) * 264 + kk * 32 + bbk];
      v16h bf = combine16(*(const v8h*)bp, *(const v8h*)(bp + 16));
      acc = __builtin_amdgcn_wmma_f32_16x16x32_f16(false, a, false, bf,
                                                   (short)0, acc, false, false);
    }
#pragma unroll
    for (int r = 0; r < 8; ++r) {
      int m = r + ((lane >> 4) << 3);
      attn[(b * LL + i0 + m) * DD + h * HD + nt * 16 + m16] = acc[r];
    }
  }
}

// ---------------------------------------------------------------------------
// attn (time part, second bandwidth monster): attn[b,i,d] += sum_j w * tv.
// Block per (b,i); lane owns 4 d's (head = lane>=16); j unrolled x4 (5
// independent loads in flight). Softmax weights past nv are exactly 0, so
// the x4 weight vector load needs no tail guard.
// ---------------------------------------------------------------------------
__global__ __launch_bounds__(256) void time_out(
    const float* __restrict__ wF, const float* __restrict__ tmv,
    const unsigned char* __restrict__ mask, float* __restrict__ attn) {
  __shared__ float accsh[128];
  __shared__ int nvsh;
  const int b = blockIdx.x / LL, i = blockIdx.x % LL;
  const int tid = threadIdx.x, wave = tid >> 5, lane = tid & 31;
  if (mask[b * LL + i]) return;                       // padded query: w row is 0
  if (tid < 128) accsh[tid] = 0.f;
  const int nv = first_padded(mask, b, tid, &nvsh);   // includes __syncthreads
  const int h = lane >> 4;
  const long rowbase = ((long)(b * LL + i)) * LL;
  const long wbase = ((long)((b * HH + h) * LL + i)) * LL;
  float4 a = make_float4(0.f, 0.f, 0.f, 0.f);
  for (int jb = wave * 4; jb < nv; jb += 32) {
    const float4 wv = *(const float4*)(wF + wbase + jb);   // w==0 past nv
    float4 t[4];
#pragma unroll
    for (int u = 0; u < 4; ++u)
      t[u] = *(const float4*)(tmv + (rowbase + jb + u) * DD + lane * 4);
#pragma unroll
    for (int u = 0; u < 4; ++u) {
      const float ww = (&wv.x)[u];
      a.x += ww * t[u].x; a.y += ww * t[u].y;
      a.z += ww * t[u].z; a.w += ww * t[u].w;
    }
  }
  atomicAdd(&accsh[lane * 4 + 0], a.x);
  atomicAdd(&accsh[lane * 4 + 1], a.y);
  atomicAdd(&accsh[lane * 4 + 2], a.z);
  atomicAdd(&accsh[lane * 4 + 3], a.w);
  __syncthreads();
  if (tid < 128) attn[(b * LL + i) * DD + tid] += accsh[tid];
}

// ---------------------------------------------------------------------------
extern "C" void kernel_launch(void* const* d_in, const int* in_sizes, int n_in,
                              void* d_out, int out_size, void* d_ws, size_t ws_size,
                              hipStream_t stream) {
  (void)in_sizes; (void)n_in; (void)out_size; (void)ws_size;
  const float* x    = (const float*)d_in[0];
  const float* apk  = (const float*)d_in[1];
  const float* apv  = (const float*)d_in[2];
  const float* tmk  = (const float*)d_in[3];
  const float* tmv  = (const float*)d_in[4];
  const unsigned char* mask = (const unsigned char*)d_in[5];   // bool mask
  const float* Wq = (const float*)d_in[6];  const float* bq = (const float*)d_in[7];
  const float* Wk = (const float*)d_in[8];  const float* bk = (const float*)d_in[9];
  const float* Wv = (const float*)d_in[10]; const float* bv = (const float*)d_in[11];
  const float* Wo = (const float*)d_in[12]; const float* bo = (const float*)d_in[13];
  float* out = (float*)d_out;

  char* ws = (char*)d_ws;
  float*    qF   = (float*)(ws);                       // 2048*128 f32 = 1 MiB
  _Float16* qH   = (_Float16*)(ws + 1048576);          // 512 KiB
  _Float16* kpH  = (_Float16*)(ws + 1572864);          // 512 KiB
  _Float16* vpH  = (_Float16*)(ws + 2097152);          // 512 KiB
  float*    wF   = (float*)(ws + 2621440);             // 8*2*256*256 f32 = 4 MiB
  _Float16* wH   = (_Float16*)(ws + 6815744);          // 2 MiB
  float*    attn = (float*)(ws + 8912896);             // 1 MiB

  // q = x@Wq+bq ; kp = x@Wk+bk+abs_pos_k ; vp = x@Wv+bv+abs_pos_v
  gemm_proj<<<128, 256, 0, stream>>>(x, Wq, bq, nullptr, qF, qH);
  gemm_proj<<<128, 256, 0, stream>>>(x, Wk, bk, apk, nullptr, kpH);
  gemm_proj<<<128, 256, 0, stream>>>(x, Wv, bv, apv, nullptr, vpH);
  // scores = q.kp^T (WMMA), then += q.tk (streaming), softmax
  scores_qk<<<256, 256, 0, stream>>>(qH, kpH, wF);
  time_scores<<<BB * LL, 256, 0, stream>>>(qF, tmk, mask, wF);
  softmax_row<<<BB * HH * LL, 256, 0, stream>>>(mask, wF, wH);
  // out = w.vp (WMMA) + w.tv (streaming), then @Wo + bo
  attn_v<<<32, 256, 0, stream>>>(wH, vpH, attn);
  time_out<<<BB * LL, 256, 0, stream>>>(wF, tmv, mask, attn);
  gemm_proj<<<128, 256, 0, stream>>>(attn, Wo, bo, nullptr, out, nullptr);
}
